// KNNAttention_22316650070219
// MI455X (gfx1250) — matrix operand order repaired and checked by reading, not compiled, not run on hardware
//
#include <hip/hip_runtime.h>
#include <math.h>

typedef __attribute__((ext_vector_type(16))) _Float16 v16h;
typedef __attribute__((ext_vector_type(8)))  _Float16 v8h;
typedef __attribute__((ext_vector_type(4)))  _Float16 v4h;
typedef __attribute__((ext_vector_type(2)))  _Float16 v2h;
typedef __attribute__((ext_vector_type(8)))  float    v8f;
typedef __attribute__((ext_vector_type(4)))  int      v4i;

__device__ __forceinline__ v8f wmma16x16x32(v16h a, v16h b, v8f c) {
  // D = A(16x32 f16) * B(32x16 f16) + C(16x16 f32)
  return __builtin_amdgcn_wmma_f32_16x16x32_f16(false, a, false, b, (short)0, c, false, false);
}

// ---------------------------------------------------------------------------
// 16-byte global->LDS staging: CDNA5 async copy (ASYNCcnt) when available,
// synchronous uint4 copy otherwise. All call sites are 16B-aligned.
// ---------------------------------------------------------------------------
#if defined(__HIPCC__) && __has_builtin(__builtin_amdgcn_global_load_async_to_lds_b128) && \
    __has_builtin(__builtin_amdgcn_s_wait_asynccnt)
#define USE_ASYNC 1
#else
#define USE_ASYNC 0
#endif

__device__ __forceinline__ void stage16(const _Float16* g, _Float16* l) {
#if USE_ASYNC
  __builtin_amdgcn_global_load_async_to_lds_b128(
      (__attribute__((address_space(1))) v4i*)(g),
      (__attribute__((address_space(3))) v4i*)(l), 0, 0);
#else
  *(uint4*)l = *(const uint4*)g;
#endif
}

__device__ __forceinline__ void stage_wait() {
#if USE_ASYNC
  __builtin_amdgcn_s_wait_asynccnt(0);
#endif
}

// ---------------------------------------------------------------------------
// f32 -> f16 conversion with optional scale (vectorized x4)
// ---------------------------------------------------------------------------
__global__ __launch_bounds__(256)
void cvt_f16_kernel(const float* __restrict__ src, _Float16* __restrict__ dst,
                    float scale, int n4) {
  int i = blockIdx.x * 256 + threadIdx.x;
  if (i < n4) {
    float4 f = ((const float4*)src)[i];
    v4h h = { (_Float16)(f.x * scale), (_Float16)(f.y * scale),
              (_Float16)(f.z * scale), (_Float16)(f.w * scale) };
    ((v4h*)dst)[i] = h;
  }
}

// ---------------------------------------------------------------------------
// GEMM: C[M,N] = A[M,K] * B[N,K]^T ; f16 operands in memory, f32 accumulate.
// Block = 128 threads (4 waves); tile 64x64; K step 32; async b128 staging.
// ---------------------------------------------------------------------------
#define TBM 64
#define TBN 64
#define TBK 32
#define GLD 40   // LDS row stride in halves: 80B, multiple of 16B

template <bool OUT16>
__global__ __launch_bounds__(128)
void gemm16_kernel(const _Float16* __restrict__ A, const _Float16* __restrict__ B,
                   void* __restrict__ Cp, int N, int K) {
  __shared__ __align__(16) _Float16 As[TBM][GLD];
  __shared__ __align__(16) _Float16 Bs[TBN][GLD];
  const int tid  = threadIdx.x;
  const int wave = tid >> 5;
  const int lane = tid & 31;
  const int half = lane >> 4;
  const int l16  = lane & 15;
  const int m0 = blockIdx.x * TBM;
  const int n0 = blockIdx.y * TBN;

  v8f acc[4] = {};

  for (int k0 = 0; k0 < K; k0 += TBK) {
    __syncthreads();
    #pragma unroll
    for (int it = 0; it < 2; ++it) {
      int i   = tid + it * 128;   // 0..255
      int r   = i >> 2;
      int c16 = i & 3;
      stage16(A + (size_t)(m0 + r) * K + k0 + c16 * 8, &As[r][c16 * 8]);
      stage16(B + (size_t)(n0 + r) * K + k0 + c16 * 8, &Bs[r][c16 * 8]);
    }
    stage_wait();
    __syncthreads();

    v16h a;
    #pragma unroll
    for (int v = 0; v < 8; ++v) {
      int kb = 2*v + 16*half;
      a[2*v]   = As[wave*16 + l16][kb];
      a[2*v+1] = As[wave*16 + l16][kb+1];
    }
    #pragma unroll
    for (int nb = 0; nb < 4; ++nb) {
      v16h bf;
      #pragma unroll
      for (int e = 0; e < 16; ++e)
        bf[e] = Bs[nb*16 + l16][16*half + e];
      acc[nb] = wmma16x16x32(a, bf, acc[nb]);
    }
  }

  #pragma unroll
  for (int nb = 0; nb < 4; ++nb)
    #pragma unroll
    for (int r = 0; r < 8; ++r) {
      size_t o = (size_t)(m0 + wave*16 + r + 8*half) * N + n0 + nb*16 + l16;
      if constexpr (OUT16) ((_Float16*)Cp)[o] = (_Float16)acc[nb][r];
      else                 ((float*)Cp)[o]    = acc[nb][r];
    }
}

// ---------------------------------------------------------------------------
// L2 norm over the sequence axis: kvp16[b,l,c] (c in 0..127) -> kn16/vn16
// ---------------------------------------------------------------------------
__global__ __launch_bounds__(256)
void l2norm_kernel(const _Float16* __restrict__ kvp, _Float16* __restrict__ kn,
                   _Float16* __restrict__ vn) {
  const int b = blockIdx.x >> 7;
  const int c = blockIdx.x & 127;
  const _Float16* src = kvp + (size_t)b * 1024 * 128 + c;
  float s = 0.f;
  for (int l = threadIdx.x; l < 1024; l += 256) {
    float x = (float)src[(size_t)l * 128];
    s += x * x;
  }
  #pragma unroll
  for (int m = 16; m >= 1; m >>= 1) s += __shfl_xor(s, m, 32);
  __shared__ float red[8];
  __shared__ float s_inv;
  if ((threadIdx.x & 31) == 0) red[threadIdx.x >> 5] = s;
  __syncthreads();
  if (threadIdx.x == 0) {
    float t = 0.f;
    #pragma unroll
    for (int i = 0; i < 8; ++i) t += red[i];
    s_inv = 1.f / fmaxf(sqrtf(t), 1e-12f);
  }
  __syncthreads();
  const float inv = s_inv;
  _Float16* dst = (c < 64) ? (kn + (size_t)b*1024*64 + c)
                           : (vn + (size_t)b*1024*64 + (c - 64));
  for (int l = threadIdx.x; l < 1024; l += 256)
    dst[(size_t)l * 64] = (_Float16)((float)src[(size_t)l * 128] * inv);
}

// ---------------------------------------------------------------------------
// Flash attention core (transposed-score formulation).
//   S^T(32k x 16q) = K_block * Q^T : K rows are the A operand, Q the B operand
//   -> per-lane (query-column) softmax stats: ONE shfl_xor(16) per reduction.
//   After exp, the S^T C-fragment IS the A-fragment of P (same key set per
//   lane) -> pure register repack, no LDS round trip for P.
//   V is staged transposed (Vt[c][key], pair-packed b32 stores) so the P*V
//   B-fragment reads are contiguous.
// qp16 is pre-scaled by 1/sqrt(dh) (folded into Wq); argmax is unaffected.
// ---------------------------------------------------------------------------
#define KS_S 72   // 144B rows, multiple of 16B

__global__ __launch_bounds__(128)
void flash_local_kernel(const _Float16* __restrict__ qp, const _Float16* __restrict__ kn,
                        const _Float16* __restrict__ vn, float* __restrict__ local_out,
                        int* __restrict__ idx_out) {
  __shared__ __align__(16) _Float16 Ks[32][KS_S];
  __shared__ __align__(16) _Float16 Vt[64][GLD];      // transposed V block
  __shared__ __align__(16) _Float16 Qs[4][16][KS_S];

  const int tid = threadIdx.x, wave = tid >> 5, lane = tid & 31;
  const int half = lane >> 4, l16 = lane & 15;
  const int qtile = blockIdx.x * 4 + wave;
  const int b   = qtile >> 10;
  const int rem = qtile & 1023;
  const int h   = rem >> 6;
  const int q0  = (rem & 63) << 4;

  // stage Q tile: 16 rows x 8 b128 chunks per wave
  for (int i = lane; i < 16 * 8; i += 32) {
    int r = i >> 3, c16 = i & 7;
    stage16(qp + (size_t)(b*1024 + q0 + r) * 1024 + h*64 + c16*8, &Qs[wave][r][c16*8]);
  }
  stage_wait();
  __syncthreads();
  // Q^T B-fragments: lane n = query = l16; element e -> c = kc*32 + 16*half + e
  v16h bq[2];
  #pragma unroll
  for (int kc = 0; kc < 2; ++kc)
    #pragma unroll
    for (int e = 0; e < 16; ++e)
      bq[kc][e] = Qs[wave][l16][kc*32 + 16*half + e];

  float m_run = -__builtin_inff(), l_run = 0.f;
  float am_v  = -__builtin_inff();
  int   am_i  = 0;
  v8f   o[4] = {};

  for (int kb0 = 0; kb0 < 1024; kb0 += 32) {
    __syncthreads();
    // K block (async): 32 rows x 8 chunks = 256 chunks / 128 threads
    #pragma unroll
    for (int it = 0; it < 2; ++it) {
      int i   = tid + it * 128;
      int r   = i >> 3;
      int c16 = i & 7;
      stage16(kn + (size_t)(b*1024 + kb0 + r) * 64 + c16*8, &Ks[r][c16*8]);
    }
    // V block transposed: thread handles key pair 2kp,2kp+1 x 8-half c chunk
    {
      int kp = tid >> 3, c8 = tid & 7;
      v8h g0 = *(const v8h*)(vn + (size_t)(b*1024 + kb0 + 2*kp    ) * 64 + c8*8);
      v8h g1 = *(const v8h*)(vn + (size_t)(b*1024 + kb0 + 2*kp + 1) * 64 + c8*8);
      #pragma unroll
      for (int e = 0; e < 8; ++e) {
        v2h pr = { g0[e], g1[e] };
        *(v2h*)&Vt[c8*8 + e][2*kp] = pr;
      }
    }
    stage_wait();
    __syncthreads();

    // S^T: A = K rows (m = key-in-group = l16), B = Q^T
    v8f s[2];
    #pragma unroll
    for (int g = 0; g < 2; ++g) {
      v8f sc = {};
      #pragma unroll
      for (int kc = 0; kc < 2; ++kc) {
        v16h ak;
        #pragma unroll
        for (int v = 0; v < 8; ++v) {
          int kb = kc*32 + 2*v + 16*half;
          ak[2*v]   = Ks[g*16 + l16][kb];
          ak[2*v+1] = Ks[g*16 + l16][kb+1];
        }
        sc = wmma16x16x32(ak, bq[kc], sc);
      }
      s[g] = sc;   // lane holds keys kb0 + g*16 + 8*half + r for query l16
    }

    // per-lane (per-query) stats over 16 local keys + one xor-16 combine
    float bm = s[0][0];
    #pragma unroll
    for (int g = 0; g < 2; ++g)
      #pragma unroll
      for (int r = 0; r < 8; ++r) bm = fmaxf(bm, s[g][r]);
    float cv = -__builtin_inff(); int ci = 0;
    #pragma unroll
    for (int g = 0; g < 2; ++g)
      #pragma unroll
      for (int r = 0; r < 8; ++r) {
        int key = kb0 + g*16 + 8*half + r;
        float v = s[g][r];
        if (v > cv || (v == cv && key < ci)) { cv = v; ci = key; }
      }
    bm = fmaxf(bm, __shfl_xor(bm, 16, 32));
    { float ov = __shfl_xor(cv, 16, 32); int oi = __shfl_xor(ci, 16, 32);
      if (ov > cv || (ov == cv && oi < ci)) { cv = ov; ci = oi; } }
    if (cv > am_v || (cv == am_v && ci < am_i)) { am_v = cv; am_i = ci; }

    float mnew  = fmaxf(m_run, bm);
    float alpha = __expf(m_run - mnew);
    m_run = mnew;
    float p[2][8], ps = 0.f;
    #pragma unroll
    for (int g = 0; g < 2; ++g)
      #pragma unroll
      for (int r = 0; r < 8; ++r) { p[g][r] = __expf(s[g][r] - mnew); ps += p[g][r]; }
    ps += __shfl_xor(ps, 16, 32);
    l_run = l_run * alpha + ps;

    // rescale O rows (row m = r + 8*half is query m; alpha lives in lane m)
    #pragma unroll
    for (int r = 0; r < 8; ++r) {
      float ar = __shfl(alpha, r + 8*half, 32);
      #pragma unroll
      for (int nb = 0; nb < 4; ++nb) o[nb][r] *= ar;
    }

    // P A-fragment straight from registers (same key set as S^T C-fragment)
    v16h pa;
    #pragma unroll
    for (int v = 0; v < 8; ++v) {
      if (v < 4) { pa[2*v] = (_Float16)p[0][2*v];     pa[2*v+1] = (_Float16)p[0][2*v+1]; }
      else       { pa[2*v] = (_Float16)p[1][2*(v-4)]; pa[2*v+1] = (_Float16)p[1][2*(v-4)+1]; }
    }
    // O(16q x 64c) += P(16x32) * V(32x64); B from transposed Vt (contiguous)
    #pragma unroll
    for (int nb = 0; nb < 4; ++nb) {
      v16h bv;
      #pragma unroll
      for (int e = 0; e < 16; ++e)
        bv[e] = Vt[nb*16 + l16][8*half + e + (e & 8)];
      o[nb] = wmma16x16x32(pa, bv, o[nb]);
    }
  }

  #pragma unroll
  for (int r = 0; r < 8; ++r) {
    float inv  = 1.f / __shfl(l_run, r + 8*half, 32);
    int   qrow = q0 + r + 8*half;
    #pragma unroll
    for (int nb = 0; nb < 4; ++nb)
      local_out[(size_t)((b*16 + h)*1024 + qrow) * 64 + nb*16 + l16] = o[nb][r] * inv;
  }
  if (half == 0) idx_out[(size_t)(b*16 + h)*1024 + q0 + l16] = am_i;
}

// ---------------------------------------------------------------------------
// Flash pass 2: retrieval attention over gathered keys/values + gate blend.
// All 4 waves share (b,h) so the gathered K/V staging is shared.
// ---------------------------------------------------------------------------
__global__ __launch_bounds__(128)
void flash_retr_kernel(const _Float16* __restrict__ qp, const _Float16* __restrict__ kn,
                       const _Float16* __restrict__ vn, const int* __restrict__ idx,
                       const float* __restrict__ local_in, const float* __restrict__ bias,
                       _Float16* __restrict__ combined) {
  __shared__ __align__(16) _Float16 Ks[32][KS_S];
  __shared__ __align__(16) _Float16 Vt[64][GLD];
  __shared__ __align__(16) _Float16 Qs[4][16][KS_S];

  const int tid = threadIdx.x, wave = tid >> 5, lane = tid & 31;
  const int half = lane >> 4, l16 = lane & 15;
  const int bh = blockIdx.x >> 4;    // 0..63
  const int b  = bh >> 4;
  const int h  = bh & 15;
  const int q0 = ((blockIdx.x & 15) * 4 + wave) * 16;

  for (int i = lane; i < 16 * 8; i += 32) {
    int r = i >> 3, c16 = i & 7;
    stage16(qp + (size_t)(b*1024 + q0 + r) * 1024 + h*64 + c16*8, &Qs[wave][r][c16*8]);
  }
  stage_wait();
  __syncthreads();
  v16h bq[2];
  #pragma unroll
  for (int kc = 0; kc < 2; ++kc)
    #pragma unroll
    for (int e = 0; e < 16; ++e)
      bq[kc][e] = Qs[wave][l16][kc*32 + 16*half + e];

  float m_run = -__builtin_inff(), l_run = 0.f;
  v8f   o[4] = {};

  for (int kb0 = 0; kb0 < 1024; kb0 += 32) {
    __syncthreads();
    #pragma unroll
    for (int it = 0; it < 2; ++it) {
      int i   = tid + it * 128;
      int r   = i >> 3;
      int c16 = i & 7;
      int j   = idx[(size_t)(b*16 + h)*1024 + kb0 + r];   // gathered key row
      stage16(kn + (size_t)(b*1024 + j) * 64 + c16*8, &Ks[r][c16*8]);
    }
    {
      int kp = tid >> 3, c8 = tid & 7;
      int j0 = idx[(size_t)(b*16 + h)*1024 + kb0 + 2*kp];
      int j1 = idx[(size_t)(b*16 + h)*1024 + kb0 + 2*kp + 1];
      v8h g0 = *(const v8h*)(vn + (size_t)(b*1024 + j0) * 64 + c8*8);
      v8h g1 = *(const v8h*)(vn + (size_t)(b*1024 + j1) * 64 + c8*8);
      #pragma unroll
      for (int e = 0; e < 8; ++e) {
        v2h pr = { g0[e], g1[e] };
        *(v2h*)&Vt[c8*8 + e][2*kp] = pr;
      }
    }
    stage_wait();
    __syncthreads();

    v8f s[2];
    #pragma unroll
    for (int g = 0; g < 2; ++g) {
      v8f sc = {};
      #pragma unroll
      for (int kc = 0; kc < 2; ++kc) {
        v16h ak;
        #pragma unroll
        for (int v = 0; v < 8; ++v) {
          int kb = kc*32 + 2*v + 16*half;
          ak[2*v]   = Ks[g*16 + l16][kb];
          ak[2*v+1] = Ks[g*16 + l16][kb+1];
        }
        sc = wmma16x16x32(ak, bq[kc], sc);
      }
      s[g] = sc;
    }

    float bm = s[0][0];
    #pragma unroll
    for (int g = 0; g < 2; ++g)
      #pragma unroll
      for (int r = 0; r < 8; ++r) bm = fmaxf(bm, s[g][r]);
    bm = fmaxf(bm, __shfl_xor(bm, 16, 32));

    float mnew  = fmaxf(m_run, bm);
    float alpha = __expf(m_run - mnew);
    m_run = mnew;
    float p[2][8], ps = 0.f;
    #pragma unroll
    for (int g = 0; g < 2; ++g)
      #pragma unroll
      for (int r = 0; r < 8; ++r) { p[g][r] = __expf(s[g][r] - mnew); ps += p[g][r]; }
    ps += __shfl_xor(ps, 16, 32);
    l_run = l_run * alpha + ps;

    #pragma unroll
    for (int r = 0; r < 8; ++r) {
      float ar = __shfl(alpha, r + 8*half, 32);
      #pragma unroll
      for (int nb = 0; nb < 4; ++nb) o[nb][r] *= ar;
    }

    v16h pa;
    #pragma unroll
    for (int v = 0; v < 8; ++v) {
      if (v < 4) { pa[2*v] = (_Float16)p[0][2*v];     pa[2*v+1] = (_Float16)p[0][2*v+1]; }
      else       { pa[2*v] = (_Float16)p[1][2*(v-4)]; pa[2*v+1] = (_Float16)p[1][2*(v-4)+1]; }
    }
    #pragma unroll
    for (int nb = 0; nb < 4; ++nb) {
      v16h bv;
      #pragma unroll
      for (int e = 0; e < 16; ++e)
        bv[e] = Vt[nb*16 + l16][8*half + e + (e & 8)];
      o[nb] = wmma16x16x32(pa, bv, o[nb]);
    }
  }

  #pragma unroll
  for (int r = 0; r < 8; ++r) {
    float inv  = 1.f / __shfl(l_run, r + 8*half, 32);
    int   qrow = q0 + r + 8*half;
    #pragma unroll
    for (int nb = 0; nb < 4; ++nb) {
      int   c   = nb*16 + l16;
      float g   = 1.f / (1.f + __expf(-bias[c]));
      float loc = local_in[(size_t)((b*16 + h)*1024 + qrow) * 64 + c];
      combined[(size_t)(b*1024 + qrow) * 1024 + h*64 + c] =
          (_Float16)(o[nb][r] * inv * g + loc * (1.f - g));
    }
  }
}

// ---------------------------------------------------------------------------
extern "C" void kernel_launch(void* const* d_in, const int* in_sizes, int n_in,
                              void* d_out, int out_size, void* d_ws, size_t ws_size,
                              hipStream_t stream) {
  (void)in_sizes; (void)n_in; (void)out_size; (void)ws_size;
  const float* q    = (const float*)d_in[0];
  const float* kv   = (const float*)d_in[1];
  const float* Wq   = (const float*)d_in[2];
  const float* Wkv  = (const float*)d_in[3];
  const float* Wc   = (const float*)d_in[4];
  const float* bias = (const float*)d_in[5];

  char* w = (char*)d_ws;
  auto alloc = [&](size_t bytes) { char* p = w; w += (bytes + 255) & ~(size_t)255; return p; };

  _Float16* q16   = (_Float16*)alloc((size_t)4096 * 1024 * 2);
  _Float16* kv16  = (_Float16*)alloc((size_t)4096 * 1024 * 2);
  _Float16* Wq16  = (_Float16*)alloc((size_t)1024 * 1024 * 2);
  _Float16* Wkv16 = (_Float16*)alloc((size_t)128  * 1024 * 2);
  _Float16* Wc16  = (_Float16*)alloc((size_t)1024 * 1024 * 2);
  _Float16* qp16  = (_Float16*)alloc((size_t)4096 * 1024 * 2);
  _Float16* kvp16 = (_Float16*)alloc((size_t)4096 * 128  * 2);
  _Float16* kn16  = (_Float16*)alloc((size_t)4 * 1024 * 64 * 2);
  _Float16* vn16  = (_Float16*)alloc((size_t)4 * 1024 * 64 * 2);
  float*    loc   = (float*)   alloc((size_t)4 * 16 * 1024 * 64 * 4);
  _Float16* cmb16 = (_Float16*)alloc((size_t)4096 * 1024 * 2);
  int*      idx   = (int*)     alloc((size_t)4 * 16 * 1024 * 4);

  // f32 -> f16 operand conversion (1/sqrt(dh) folded into Wq)
  cvt_f16_kernel<<<4096, 256, 0, stream>>>(q,   q16,   1.0f,   1048576);
  cvt_f16_kernel<<<4096, 256, 0, stream>>>(kv,  kv16,  1.0f,   1048576);
  cvt_f16_kernel<<<1024, 256, 0, stream>>>(Wq,  Wq16,  0.125f, 262144);
  cvt_f16_kernel<<<128,  256, 0, stream>>>(Wkv, Wkv16, 1.0f,   32768);
  cvt_f16_kernel<<<1024, 256, 0, stream>>>(Wc,  Wc16,  1.0f,   262144);

  // qp = (q @ Wq^T) * 0.125 ; kvp = kv @ Wkv^T
  gemm16_kernel<true><<<dim3(64, 16), 128, 0, stream>>>(q16,  Wq16,  qp16,  1024, 1024);
  gemm16_kernel<true><<<dim3(64, 2),  128, 0, stream>>>(kv16, Wkv16, kvp16, 128,  1024);
  // sequence-axis L2 norm -> kn16, vn16
  l2norm_kernel<<<512, 256, 0, stream>>>(kvp16, kn16, vn16);
  // local attention + argmax indices
  flash_local_kernel<<<1024, 128, 0, stream>>>(qp16, kn16, vn16, loc, idx);
  // retrieval attention over gathered keys + gate blend -> (b,l,d) f16
  flash_retr_kernel<<<1024, 128, 0, stream>>>(qp16, kn16, vn16, idx, loc, bias, cmb16);
  // out = combined @ Wc^T  (f32 output)
  gemm16_kernel<false><<<dim3(64, 16), 128, 0, stream>>>(cmb16, Wc16, (float*)d_out, 1024, 1024);
}
